// CascadeEncoder_70712341561347
// MI455X (gfx1250) — matrix operand order repaired and checked by reading, not compiled, not run on hardware
//
#include <hip/hip_runtime.h>
#include <hip/hip_bf16.h>
#include <stdint.h>

// ---------------------------------------------------------------------------
// Types for CDNA5 WMMA (wave32, v_wmma_f32_16x16x32_bf16)
// ---------------------------------------------------------------------------
typedef __bf16 bf16_t;
typedef __attribute__((ext_vector_type(8)))  __bf16 v8bf;
typedef __attribute__((ext_vector_type(16))) __bf16 v16bf;
typedef __attribute__((ext_vector_type(8)))  float  v8f;

union V16U { v16bf v; v8bf h[2]; };

static __device__ __forceinline__ v16bf load_frag16(const bf16_t* p0, const bf16_t* p1) {
  V16U u;
  u.h[0] = *(const v8bf*)p0;
  u.h[1] = *(const v8bf*)p1;
  return u.v;
}

static __device__ __forceinline__ v8f wmma_bf16(v16bf a, v16bf b, v8f c) {
  // (neg_a, A, neg_b, B, c_mod, C, reuse_a, reuse_b)
  return __builtin_amdgcn_wmma_f32_16x16x32_bf16(false, a, false, b, (short)0, c, false, false);
}

static __device__ __forceinline__ bf16_t f2bf(float f) {
  union { float f; uint32_t u; } in; in.f = f;
  uint32_t u = in.u;
  uint32_t r = (u + 0x7FFFu + ((u >> 16) & 1u)) >> 16;   // RNE
  union { uint16_t s; bf16_t b; } out; out.s = (uint16_t)r;
  return out.b;
}

static __device__ __forceinline__ float bf2f(bf16_t b) {
  union { uint16_t s; bf16_t b; } in; in.b = b;
  union { uint32_t u; float f; } out; out.u = ((uint32_t)in.s) << 16;
  return out.f;
}

// Async global -> LDS copy of 16B per lane (+ a second 16B at +16 in both
// address spaces). Tracked by ASYNCcnt; waiter below.
static __device__ __forceinline__ void async_copy_b128x2(void* lds_ptr, const void* gptr) {
  uint32_t la = (uint32_t)(uintptr_t)lds_ptr;   // LDS byte offset lives in addr[31:0]
  asm volatile("global_load_async_to_lds_b128 %0, %1, off"
               :: "v"(la), "v"(gptr) : "memory");
  asm volatile("global_load_async_to_lds_b128 %0, %1, off offset:16"
               :: "v"(la), "v"(gptr) : "memory");
}
static __device__ __forceinline__ void wait_async0() {
  asm volatile("s_wait_asynccnt 0" ::: "memory");
}

// Problem constants
#define DIMC   512
#define LSEQ   1024
#define NH     8
#define DKV    128
#define HD     1024           // NH * DKV
#define BATCH  16
#define MROWS  (BATCH * LSEQ) // 16384
#define NEGV   (-4294967295.0f)

// ---------------------------------------------------------------------------
// 1) x = seq + pos ; keep f32 (residual) and bf16 (GEMM input)
// ---------------------------------------------------------------------------
__global__ __launch_bounds__(256) void prep_x_kernel(
    const float* __restrict__ seq, const float* __restrict__ pos,
    float* __restrict__ xf, bf16_t* __restrict__ xb)
{
  size_t i = (size_t)blockIdx.x * 256 + threadIdx.x;   // over B*L*D
  int d = (int)(i % DIMC);
  int l = (int)((i / DIMC) % LSEQ);
  float v = seq[i] + pos[(size_t)l * DIMC + d];
  xf[i] = v;
  xb[i] = f2bf(v);
}

// ---------------------------------------------------------------------------
// 2) Weight convert (optionally transposed) fp32 -> bf16, dst is Bt [N][K]
// ---------------------------------------------------------------------------
__global__ __launch_bounds__(256) void cvt_w_kernel(
    const float* __restrict__ src, bf16_t* __restrict__ dst,
    int R, int C, int transpose)
{
  size_t i = (size_t)blockIdx.x * 256 + threadIdx.x;
  if (i >= (size_t)R * C) return;
  int r = (int)(i / C), c = (int)(i % C);
  bf16_t v = f2bf(src[i]);
  if (transpose) dst[(size_t)c * R + r] = v;
  else           dst[i] = v;
}

// ---------------------------------------------------------------------------
// 3) Generic bf16 WMMA GEMM:  C[M,N] = A[M,K] * Bt[N,K]^T  (+bias)(+relu)
//    128x128 block tile, 8 waves (4Mx2N), K-step 32.
//    Double-buffered LDS fed by GLOBAL_LOAD_ASYNC_TO_LDS_B128; one barrier
//    per K-step; async tile i+1 streams while WMMA consumes tile i.
// ---------------------------------------------------------------------------
__global__ __launch_bounds__(256) void gemm_bf16_kernel(
    const bf16_t* __restrict__ A, const bf16_t* __restrict__ Bt,
    float* __restrict__ Cf, bf16_t* __restrict__ Cb,
    const float* __restrict__ bias,
    int M, int N, int K, int lda, int ldb, int ldc, int relu)
{
  __shared__ __align__(16) bf16_t As[2][128 * 40];
  __shared__ __align__(16) bf16_t Bs[2][128 * 40];

  const int tid  = threadIdx.x;
  const int lane = tid & 31;
  const int wid  = tid >> 5;
  const int wm   = wid >> 1;          // 0..3
  const int wn   = wid & 1;           // 0..1
  const int mloc = lane & 15;
  const int hh   = lane >> 4;         // lane half
  const int bm   = blockIdx.y * 128;
  const int bn   = blockIdx.x * 128;

  const int srow  = tid >> 1;         // staging row 0..127
  const int shalf = (tid & 1) * 16;   // staging k half

  v8f acc[2][4];
  for (int s = 0; s < 2; ++s)
    for (int t = 0; t < 4; ++t)
      acc[s][t] = v8f{};

  auto stage = [&](int k0, int buf) {
    async_copy_b128x2(&As[buf][srow * 40 + shalf],
                      A  + (size_t)(bm + srow) * lda + k0 + shalf);
    async_copy_b128x2(&Bs[buf][srow * 40 + shalf],
                      Bt + (size_t)(bn + srow) * ldb + k0 + shalf);
  };

  const int nIter = K >> 5;
  stage(0, 0);
  int buf = 0;
  for (int it = 0; it < nIter; ++it) {
    wait_async0();                 // my staged chunks for tile `it` landed
    __syncthreads();               // everyone's landed; prior reads retired
    if (it + 1 < nIter) stage((it + 1) << 5, buf ^ 1);

    v16bf af[2], bfv[4];
    for (int s = 0; s < 2; ++s) {
      const bf16_t* base = As[buf] + (size_t)(wm * 32 + s * 16 + mloc) * 40;
      af[s] = load_frag16(base + 8 * hh, base + 16 + 8 * hh);
    }
    for (int t = 0; t < 4; ++t) {
      const bf16_t* base = Bs[buf] + (size_t)(wn * 64 + t * 16 + mloc) * 40;
      bfv[t] = load_frag16(base + 8 * hh, base + 16 + 8 * hh);
    }
    for (int s = 0; s < 2; ++s)
      for (int t = 0; t < 4; ++t)
        acc[s][t] = wmma_bf16(af[s], bfv[t], acc[s][t]);
    buf ^= 1;
  }

  for (int s = 0; s < 2; ++s)
    for (int t = 0; t < 4; ++t)
      for (int r = 0; r < 8; ++r) {
        int row = bm + wm * 32 + s * 16 + r + 8 * hh;
        int col = bn + wn * 64 + t * 16 + mloc;
        float v = acc[s][t][r];
        if (bias) v += bias[col];
        if (relu) v = fmaxf(v, 0.0f);
        if (Cf) Cf[(size_t)row * ldc + col] = v;
        if (Cb) Cb[(size_t)row * ldc + col] = f2bf(v);
      }
}

// ---------------------------------------------------------------------------
// 4) V -> Vt[bh][d][key] (bf16), LDS-tiled 32x32 transpose per (b,h)
// ---------------------------------------------------------------------------
__global__ __launch_bounds__(256) void transpose_v_kernel(
    const bf16_t* __restrict__ V, bf16_t* __restrict__ Vt)
{
  __shared__ __align__(8) bf16_t tile[32][36];
  const int bh  = blockIdx.z;
  const int b   = bh >> 3, hd = bh & 7;
  const int key0 = blockIdx.x * 32;
  const int d0   = blockIdx.y * 32;
  const int t    = threadIdx.x;
  const int row  = t >> 3;            // 0..31
  const int quad = (t & 7) * 4;       // 0,4,...,28

  const bf16_t* src = V + ((size_t)(b * LSEQ + key0 + row)) * HD + hd * DKV + d0 + quad;
  *(uint2*)&tile[row][quad] = *(const uint2*)src;
  __syncthreads();

  __align__(8) bf16_t vals[4];
  for (int e = 0; e < 4; ++e) vals[e] = tile[quad + e][row];
  bf16_t* dst = Vt + ((size_t)(bh * DKV + d0 + row)) * LSEQ + key0 + quad;
  *(uint2*)dst = *(const uint2*)vals;
}

// ---------------------------------------------------------------------------
// 5) meanV[bh][d] = mean over keys of V (for fully-padded query rows)
// ---------------------------------------------------------------------------
__global__ __launch_bounds__(128) void meanv_kernel(
    const bf16_t* __restrict__ V, float* __restrict__ meanV)
{
  int bh = blockIdx.x;            // 0..127
  int d  = threadIdx.x;           // 0..127
  int b  = bh >> 3, hd = bh & 7;
  float s = 0.0f;
  const bf16_t* base = V + ((size_t)b * LSEQ) * HD + hd * DKV + d;
  for (int k = 0; k < LSEQ; ++k) s += bf2f(base[(size_t)k * HD]);
  meanV[(size_t)bh * DKV + d] = s * (1.0f / (float)LSEQ);
}

// ---------------------------------------------------------------------------
// 6) Flash attention: block = (b, h, 128 q rows), 8 waves x 16 q rows each.
//    Both GEMMs via v_wmma_f32_16x16x32_bf16; causal tile skipping; online
//    softmax; async double-buffered K / Vt staging.
// ---------------------------------------------------------------------------
__global__ __launch_bounds__(256) void attn_kernel(
    const bf16_t* __restrict__ Q, const bf16_t* __restrict__ Kg,
    const bf16_t* __restrict__ Vtg, const unsigned char* __restrict__ pad,
    const float* __restrict__ meanV, bf16_t* __restrict__ ctx)
{
  __shared__ __align__(16) bf16_t Kt[2][32 * 136];   // [key][d]  (B of S-GEMM)
  __shared__ __align__(16) bf16_t Vt[2][128 * 40];   // [d][key]  (B of PV-GEMM)
  __shared__ __align__(16) bf16_t Ps[8 * 16 * 40];   // per-wave P panel [q][key]

  const int tid  = threadIdx.x;
  const int lane = tid & 31;
  const int wid  = tid >> 5;
  const int mloc = lane & 15;
  const int hh   = lane >> 4;
  const int bh   = blockIdx.y;
  const int b    = bh >> 3, hd = bh & 7;
  const int qbase = blockIdx.x * 128;
  const int qw    = qbase + wid * 16;

  // Q fragments (16 rows x 128 d) kept in VGPRs for the whole key loop
  v16bf qf[4];
  {
    const bf16_t* qp = Q + ((size_t)(b * LSEQ + qw + mloc)) * HD + hd * DKV;
    for (int c = 0; c < 4; ++c)
      qf[c] = load_frag16(qp + 32 * c + 8 * hh, qp + 32 * c + 16 + 8 * hh);
  }

  float mrun[8], lrun[8];
  v8f acc[8];
  for (int r = 0; r < 8; ++r) { mrun[r] = -3.0e38f; lrun[r] = 0.0f; }
  for (int t = 0; t < 8; ++t) acc[t] = v8f{};

  const float inv_temp = 1.0f / (sqrtf(128.0f) + 1e-6f);
  const int nk_block = (qbase + 128) >> 5;           // causal bound (block)
  const int nk_wave  = (qw + 16 + 31) >> 5;          // causal bound (wave)

  const int krow = tid >> 3;            // staging: key 0..31
  const int kseg = (tid & 7) * 16;      // staging: d segment
  const int vrow = tid >> 1;            // staging: d 0..127
  const int vseg = (tid & 1) * 16;      // staging: key segment

  auto stage = [&](int kt, int bufi) {
    const int kb = kt * 32;
    async_copy_b128x2(&Kt[bufi][krow * 136 + kseg],
                      Kg + ((size_t)(b * LSEQ + kb + krow)) * HD + hd * DKV + kseg);
    async_copy_b128x2(&Vt[bufi][vrow * 40 + vseg],
                      Vtg + ((size_t)(bh * DKV + vrow)) * LSEQ + kb + vseg);
  };

  stage(0, 0);
  int buf = 0;
  for (int kt = 0; kt < nk_block; ++kt) {
    wait_async0();
    __syncthreads();
    if (kt + 1 < nk_block) stage(kt + 1, buf ^ 1);

    if (kt < nk_wave) {
      const int kbase = kt * 32;
      // ---- scores: two 16x16 key tiles, contraction over d=128 (4 WMMA each)
      v8f sc[2];
      for (int t2 = 0; t2 < 2; ++t2) {
        sc[t2] = v8f{};
        for (int c = 0; c < 4; ++c) {
          const bf16_t* kb = Kt[buf] + (size_t)(t2 * 16 + mloc) * 136 + 32 * c + 8 * hh;
          v16bf kf = load_frag16(kb, kb + 16);
          sc[t2] = wmma_bf16(qf[c], kf, sc[t2]);
        }
      }
      // ---- scale + causal mask (C layout: row = r+8*hh, col = mloc)
      for (int t2 = 0; t2 < 2; ++t2) {
        int kpos = kbase + t2 * 16 + mloc;
        for (int r = 0; r < 8; ++r) {
          int qrow = qw + r + 8 * hh;
          float s = sc[t2][r] * inv_temp;
          sc[t2][r] = (kpos > qrow) ? NEGV : s;
        }
      }
      // ---- online softmax update (row reductions across 16-lane halves)
      for (int r = 0; r < 8; ++r) {
        float tm = fmaxf(sc[0][r], sc[1][r]);
        for (int off = 1; off < 16; off <<= 1)
          tm = fmaxf(tm, __shfl_xor(tm, off, 32));
        float mnew = fmaxf(mrun[r], tm);
        float fac  = __expf(mrun[r] - mnew);
        float p0   = __expf(sc[0][r] - mnew);
        float p1   = __expf(sc[1][r] - mnew);
        sc[0][r] = p0; sc[1][r] = p1;
        float ts = p0 + p1;
        for (int off = 1; off < 16; off <<= 1)
          ts += __shfl_xor(ts, off, 32);
        lrun[r] = lrun[r] * fac + ts;
        mrun[r] = mnew;
        for (int t = 0; t < 8; ++t) acc[t][r] *= fac;
      }
      // ---- P (bf16) through per-wave LDS panel -> A fragment layout
      bf16_t* pw = Ps + (size_t)wid * 16 * 40;
      for (int t2 = 0; t2 < 2; ++t2)
        for (int r = 0; r < 8; ++r)
          pw[(r + 8 * hh) * 40 + t2 * 16 + mloc] = f2bf(sc[t2][r]);
      asm volatile("s_wait_dscnt 0" ::: "memory");   // same-wave LDS RAW
      const bf16_t* pr = pw + (size_t)mloc * 40;
      v16bf pf = load_frag16(pr + 8 * hh, pr + 16 + 8 * hh);
      // ---- ctx += P @ V  (8 d-subtiles of 16)
      for (int t = 0; t < 8; ++t) {
        const bf16_t* vb = Vt[buf] + (size_t)(t * 16 + mloc) * 40 + 8 * hh;
        v16bf vf = load_frag16(vb, vb + 16);
        acc[t] = wmma_bf16(pf, vf, acc[t]);
      }
    }
    buf ^= 1;
  }

  // ---- epilogue: 1/l scaling; padded rows -> mean over all V (ref semantics)
  for (int r = 0; r < 8; ++r) {
    int qrow = qw + r + 8 * hh;
    float inv = 1.0f / lrun[r];
    bool pd = pad[(size_t)b * LSEQ + qrow] != 0;
    bf16_t* op = ctx + ((size_t)(b * LSEQ + qrow)) * HD + hd * DKV;
    for (int t = 0; t < 8; ++t) {
      int d = t * 16 + mloc;
      float v = pd ? meanV[(size_t)bh * DKV + d] : acc[t][r] * inv;
      op[d] = f2bf(v);
    }
  }
}

// ---------------------------------------------------------------------------
// 7) out = LayerNorm(A + R) ; optional f32 and bf16 outputs. One row/block.
// ---------------------------------------------------------------------------
__global__ __launch_bounds__(256) void add_ln_kernel(
    const float* __restrict__ A, const float* __restrict__ R,
    const float* __restrict__ g, const float* __restrict__ bta,
    float* __restrict__ outF, bf16_t* __restrict__ outB)
{
  __shared__ float sh1[256];
  __shared__ float sh2[256];
  const size_t row = blockIdx.x;
  const int tid = threadIdx.x;
  const size_t base = row * DIMC;
  float v0 = A[base + tid]       + R[base + tid];
  float v1 = A[base + 256 + tid] + R[base + 256 + tid];
  sh1[tid] = v0 + v1;
  sh2[tid] = v0 * v0 + v1 * v1;
  __syncthreads();
  for (int o = 128; o > 0; o >>= 1) {
    if (tid < o) { sh1[tid] += sh1[tid + o]; sh2[tid] += sh2[tid + o]; }
    __syncthreads();
  }
  float mu   = sh1[0] * (1.0f / DIMC);
  float var  = sh2[0] * (1.0f / DIMC) - mu * mu;
  float rstd = rsqrtf(var + 1e-5f);
  float o0 = (v0 - mu) * rstd * g[tid]       + bta[tid];
  float o1 = (v1 - mu) * rstd * g[tid + 256] + bta[tid + 256];
  if (outF) { outF[base + tid] = o0; outF[base + 256 + tid] = o1; }
  if (outB) { outB[base + tid] = f2bf(o0); outB[base + 256 + tid] = f2bf(o1); }
}

// ---------------------------------------------------------------------------
// Launch
// ---------------------------------------------------------------------------
extern "C" void kernel_launch(void* const* d_in, const int* in_sizes, int n_in,
                              void* d_out, int out_size, void* d_ws, size_t ws_size,
                              hipStream_t stream) {
  const float*         seq  = (const float*)d_in[0];
  const unsigned char* pad  = (const unsigned char*)d_in[1];   // jnp bool -> u8
  const float*         pos  = (const float*)d_in[2];
  const float*         W_q  = (const float*)d_in[3];
  const float*         W_k  = (const float*)d_in[4];
  const float*         W_v  = (const float*)d_in[5];
  const float*         W_o  = (const float*)d_in[6];
  const float*         w1   = (const float*)d_in[7];
  const float*         b1   = (const float*)d_in[8];
  const float*         w2   = (const float*)d_in[9];
  const float*         b2   = (const float*)d_in[10];
  const float*         ln_g = (const float*)d_in[11];
  const float*         ln_b = (const float*)d_in[12];
  float* out = (float*)d_out;

  // workspace carve-up (256B aligned)
  char* ws = (char*)d_ws;
  size_t off = 0;
  auto carve = [&](size_t bytes) -> char* {
    char* p = ws + off;
    off = (off + bytes + 255) & ~(size_t)255;
    return p;
  };
  const size_t MD  = (size_t)MROWS * DIMC;   // 16384*512
  const size_t MHD = (size_t)MROWS * HD;     // 16384*1024

  float*  xf    = (float*) carve(MD * 4);
  bf16_t* xb    = (bf16_t*)carve(MD * 2);
  bf16_t* wqT   = (bf16_t*)carve((size_t)DIMC * HD * 2);
  bf16_t* wkT   = (bf16_t*)carve((size_t)DIMC * HD * 2);
  bf16_t* wvT   = (bf16_t*)carve((size_t)DIMC * HD * 2);
  bf16_t* woT   = (bf16_t*)carve((size_t)DIMC * HD * 2);
  bf16_t* w1b   = (bf16_t*)carve((size_t)DIMC * DIMC * 2);
  bf16_t* w2b   = (bf16_t*)carve((size_t)DIMC * DIMC * 2);
  bf16_t* Qb    = (bf16_t*)carve(MHD * 2);
  bf16_t* Kb    = (bf16_t*)carve(MHD * 2);
  bf16_t* Vb    = (bf16_t*)carve(MHD * 2);
  bf16_t* Vtb   = (bf16_t*)carve(MHD * 2);   // [bh][d][key]
  bf16_t* ctxb  = (bf16_t*)carve(MHD * 2);
  float*  mv    = (float*) carve((size_t)BATCH * NH * DKV * 4);
  float*  vatt  = (float*) carve(MD * 4);
  float*  Xf    = (float*) carve(MD * 4);
  bf16_t* Xb    = (bf16_t*)carve(MD * 2);
  bf16_t* Hb    = (bf16_t*)carve(MD * 2);
  float*  Yf    = (float*) carve(MD * 4);
  (void)ws_size; (void)in_sizes; (void)n_in; (void)out_size;

  // 1) x = seq + pos
  prep_x_kernel<<<(unsigned)(MD / 256), 256, 0, stream>>>(seq, pos, xf, xb);

  // 2) weights -> bf16 Bt layout
  cvt_w_kernel<<<(unsigned)((size_t)DIMC * HD / 256), 256, 0, stream>>>(W_q, wqT, DIMC, HD, 1);
  cvt_w_kernel<<<(unsigned)((size_t)DIMC * HD / 256), 256, 0, stream>>>(W_k, wkT, DIMC, HD, 1);
  cvt_w_kernel<<<(unsigned)((size_t)DIMC * HD / 256), 256, 0, stream>>>(W_v, wvT, DIMC, HD, 1);
  cvt_w_kernel<<<(unsigned)((size_t)HD * DIMC / 256), 256, 0, stream>>>(W_o, woT, HD, DIMC, 1);
  cvt_w_kernel<<<(unsigned)((size_t)DIMC * DIMC / 256), 256, 0, stream>>>(w1, w1b, DIMC, DIMC, 0);
  cvt_w_kernel<<<(unsigned)((size_t)DIMC * DIMC / 256), 256, 0, stream>>>(w2, w2b, DIMC, DIMC, 0);

  // 3) Q,K,V projections (bf16 out)
  dim3 blk(256);
  dim3 gQKV(HD / 128, MROWS / 128);
  gemm_bf16_kernel<<<gQKV, blk, 0, stream>>>(xb, wqT, nullptr, Qb, nullptr,
                                             MROWS, HD, DIMC, DIMC, DIMC, HD, 0);
  gemm_bf16_kernel<<<gQKV, blk, 0, stream>>>(xb, wkT, nullptr, Kb, nullptr,
                                             MROWS, HD, DIMC, DIMC, DIMC, HD, 0);
  gemm_bf16_kernel<<<gQKV, blk, 0, stream>>>(xb, wvT, nullptr, Vb, nullptr,
                                             MROWS, HD, DIMC, DIMC, DIMC, HD, 0);

  // 4) V -> Vt[bh][d][key] ; 5) mean-V per (b,h)
  dim3 gT(LSEQ / 32, DKV / 32, BATCH * NH);
  transpose_v_kernel<<<gT, blk, 0, stream>>>(Vb, Vtb);
  meanv_kernel<<<BATCH * NH, 128, 0, stream>>>(Vb, mv);

  // 6) flash attention -> ctx (bf16)
  dim3 gAtt(LSEQ / 128, BATCH * NH);
  attn_kernel<<<gAtt, blk, 0, stream>>>(Qb, Kb, Vtb, pad, mv, ctxb);

  // 7) output projection (f32 out)
  dim3 gO(DIMC / 128, MROWS / 128);
  gemm_bf16_kernel<<<gO, blk, 0, stream>>>(ctxb, woT, vatt, nullptr, nullptr,
                                           MROWS, DIMC, HD, HD, HD, DIMC, 0);

  // 8) X = LN(x + v_att)
  add_ln_kernel<<<MROWS, 256, 0, stream>>>(vatt, xf, ln_g, ln_b, Xf, Xb);

  // 9) FFN1: relu(X @ w1^T + b1) (bf16 out)
  gemm_bf16_kernel<<<gO, blk, 0, stream>>>(Xb, w1b, nullptr, Hb, b1,
                                           MROWS, DIMC, DIMC, DIMC, DIMC, DIMC, 1);
  // 10) FFN2: H @ w2^T + b2 (f32 out)
  gemm_bf16_kernel<<<gO, blk, 0, stream>>>(Hb, w2b, Yf, nullptr, b2,
                                           MROWS, DIMC, DIMC, DIMC, DIMC, DIMC, 0);

  // 11) out = LN(Y + X)
  add_ln_kernel<<<MROWS, 256, 0, stream>>>(Yf, Xf, ln_g, ln_b, out, nullptr);
}
